// SpiralConvBlock_85057532330638
// MI455X (gfx1250) — matrix operand order repaired and checked by reading, not compiled, not run on hardware
//
#include <hip/hip_runtime.h>

// ---------------- problem constants (B,L,D fixed by the reference) ----------
static constexpr int Bsz  = 4;
static constexpr int Lsz  = 4096;
static constexpr int Dsz  = 512;
static constexpr int DFsz = 1024;
static constexpr int Rrows = Bsz * Lsz;        // 16384 rows for layernorm
static constexpr int NC = 16;                  // scan chunks per sequence
static constexpr int CL = Lsz / NC;            // 256 steps per chunk

// ---------------- WMMA types ------------------------------------------------
typedef __attribute__((ext_vector_type(16))) __bf16 v16bf;
typedef __attribute__((ext_vector_type(8)))  float  v8f;

union Frag { v16bf v; unsigned u[8]; };

__device__ __forceinline__ unsigned short f2bf(float f) {
  unsigned u = __float_as_uint(f);
  u += 0x7FFFu + ((u >> 16) & 1u);     // round-to-nearest-even
  return (unsigned short)(u >> 16);
}

// ---------------- LayerNorm row stats (one wave32 per row) ------------------
__global__ void ln_stats_kernel(const float* __restrict__ X,
                                float* __restrict__ mean,
                                float* __restrict__ rstd, int R) {
  const int warp = threadIdx.x >> 5;
  const int lane = threadIdx.x & 31;
  const int row  = blockIdx.x * (blockDim.x >> 5) + warp;
  if (row >= R) return;
  const float* p = X + (size_t)row * Dsz;
  float s = 0.f, sq = 0.f;
#pragma unroll
  for (int j = 0; j < Dsz / 32; ++j) {
    float t = p[lane + j * 32];
    s += t; sq += t * t;
  }
#pragma unroll
  for (int off = 16; off >= 1; off >>= 1) {
    s  += __shfl_xor(s,  off, 32);
    sq += __shfl_xor(sq, off, 32);
  }
  const float m  = s * (1.0f / Dsz);
  const float va = sq * (1.0f / Dsz) - m * m;
  if (lane == 0) { mean[row] = m; rstd[row] = rsqrtf(va + 1e-5f); }
}

// ---------------- fused LN + bf16 conversion (one wave32 per row) -----------
__global__ void ln_bf16_kernel(const float* __restrict__ X,
                               const float* __restrict__ g,
                               const float* __restrict__ bb,
                               unsigned short* __restrict__ Out, int R) {
  const int warp = threadIdx.x >> 5;
  const int lane = threadIdx.x & 31;
  const int row  = blockIdx.x * (blockDim.x >> 5) + warp;
  if (row >= R) return;
  const float* p = X + (size_t)row * Dsz;
  float v[Dsz / 32];
  float s = 0.f, sq = 0.f;
#pragma unroll
  for (int j = 0; j < Dsz / 32; ++j) {
    float t = p[lane + j * 32];
    v[j] = t; s += t; sq += t * t;
  }
#pragma unroll
  for (int off = 16; off >= 1; off >>= 1) {
    s  += __shfl_xor(s,  off, 32);
    sq += __shfl_xor(sq, off, 32);
  }
  const float m  = s * (1.0f / Dsz);
  const float rs = rsqrtf(sq * (1.0f / Dsz) - m * m + 1e-5f);
  unsigned short* o = Out + (size_t)row * Dsz;
#pragma unroll
  for (int j = 0; j < Dsz / 32; ++j) {
    const int c = lane + j * 32;
    o[c] = f2bf((v[j] - m) * rs * g[c] + bb[c]);
  }
}

// ---------------- phazor helper --------------------------------------------
__device__ __forceinline__ void load_phazor(const float* pre, const float* pim,
                                            int d, float& phr, float& phi) {
  const float pr = pre[d], pi = pim[d];
  const float ab = sqrtf(pr * pr + pi * pi);
  const float sc = __expf(-ab) / ab;   // p/|p| * exp(-|p|)
  phr = pr * sc; phi = pi * sc;
}

// ---------------- scan pass 1: per-chunk local scan -> chunk carries --------
__global__ void scan_pass1_kernel(const float* __restrict__ X,
                                  const float* __restrict__ mean,
                                  const float* __restrict__ rstd,
                                  const float* __restrict__ g,
                                  const float* __restrict__ bb,
                                  const float* __restrict__ pre,
                                  const float* __restrict__ pim,
                                  const float* __restrict__ pinr,
                                  const float* __restrict__ pini,
                                  float2* __restrict__ carryLoc) {
  const int gid  = blockIdx.x * blockDim.x + threadIdx.x;
  if (gid >= Bsz * NC * Dsz) return;
  const int d    = gid % Dsz;
  const int rest = gid / Dsz;
  const int c    = rest % NC;
  const int b    = rest / NC;
  float phr, phi; load_phazor(pre, pim, d, phr, phi);
  const float qr = pinr[d], qi = pini[d];
  const float gd = g[d], bd = bb[d];
  float hr = 0.f, hi = 0.f;
  const int t0 = c * CL;
  const float* xp = X    + ((size_t)b * Lsz + t0) * Dsz + d;
  const float* mp = mean + (size_t)b * Lsz + t0;
  const float* rp = rstd + (size_t)b * Lsz + t0;
  for (int t = 0; t < CL; ++t) {
    const float xv = xp[(size_t)t * Dsz];
    const float xn = (xv - mp[t]) * rp[t] * gd + bd;
    const float nr = phr * hr - phi * hi + qr * xn;
    const float ni = phr * hi + phi * hr + qi * xn;
    hr = nr; hi = ni;
  }
  carryLoc[((size_t)b * NC + c) * Dsz + d] = make_float2(hr, hi);
}

// ---------------- carry chain across chunks (tiny) --------------------------
__global__ void scan_carry_kernel(const float2* __restrict__ carryLoc,
                                  const float* __restrict__ h0r,
                                  const float* __restrict__ h0i,
                                  const float* __restrict__ pre,
                                  const float* __restrict__ pim,
                                  float2* __restrict__ carryIn) {
  const int gid = blockIdx.x * blockDim.x + threadIdx.x;
  if (gid >= Bsz * Dsz) return;
  const int d = gid % Dsz;
  const int b = gid / Dsz;
  float phr, phi; load_phazor(pre, pim, d, phr, phi);
  // phazor^CL via 8 complex squarings (CL = 256 = 2^8)
  float wr = phr, wi = phi;
#pragma unroll
  for (int i = 0; i < 8; ++i) {
    const float nr = wr * wr - wi * wi;
    wi = 2.0f * wr * wi;
    wr = nr;
  }
  float Hr = h0r[b * Dsz + d], Hi = h0i[b * Dsz + d];
  for (int c = 0; c < NC; ++c) {
    const size_t idx = ((size_t)b * NC + c) * Dsz + d;
    carryIn[idx] = make_float2(Hr, Hi);           // hidden at chunk start - 1
    const float2 lc = carryLoc[idx];
    const float nr = lc.x + wr * Hr - wi * Hi;
    const float ni = lc.y + wr * Hi + wi * Hr;
    Hr = nr; Hi = ni;
  }
}

// ---------------- scan pass 2: final scan, write hidden (complex) + y -------
__global__ void scan_pass2_kernel(const float* __restrict__ X,
                                  const float* __restrict__ mean,
                                  const float* __restrict__ rstd,
                                  const float* __restrict__ g,
                                  const float* __restrict__ bb,
                                  const float* __restrict__ pre,
                                  const float* __restrict__ pim,
                                  const float* __restrict__ pinr,
                                  const float* __restrict__ pini,
                                  const float2* __restrict__ carryIn,
                                  float2* __restrict__ hidOut,
                                  float* __restrict__ Y) {
  const int gid  = blockIdx.x * blockDim.x + threadIdx.x;
  if (gid >= Bsz * NC * Dsz) return;
  const int d    = gid % Dsz;
  const int rest = gid / Dsz;
  const int c    = rest % NC;
  const int b    = rest / NC;
  float phr, phi; load_phazor(pre, pim, d, phr, phi);
  const float qr = pinr[d], qi = pini[d];
  const float gd = g[d], bd = bb[d];
  const float2 H = carryIn[((size_t)b * NC + c) * Dsz + d];
  float hr = H.x, hi = H.y;
  const int t0 = c * CL;
  const float* xp = X    + ((size_t)b * Lsz + t0) * Dsz + d;
  const float* mp = mean + (size_t)b * Lsz + t0;
  const float* rp = rstd + (size_t)b * Lsz + t0;
  float2* hp = hidOut + ((size_t)b * Lsz + t0) * Dsz + d;
  float*  yp = Y      + ((size_t)b * Lsz + t0) * Dsz + d;
  for (int t = 0; t < CL; ++t) {
    const float xv = xp[(size_t)t * Dsz];
    const float xn = (xv - mp[t]) * rp[t] * gd + bd;
    const float nr = phr * hr - phi * hi + qr * xn;
    const float ni = phr * hi + phi * hr + qi * xn;
    hr = nr; hi = ni;
    hp[(size_t)t * Dsz] = make_float2(hr, hi);
    yp[(size_t)t * Dsz] = hr + xv;
  }
}

// ---------------- weight f32 -> bf16 ---------------------------------------
__global__ void wcvt_kernel(const float* __restrict__ w1,
                            const float* __restrict__ w2,
                            unsigned short* __restrict__ o1,
                            unsigned short* __restrict__ o2, int n) {
  const int i = blockIdx.x * blockDim.x + threadIdx.x;
  if (i < n)              o1[i] = f2bf(w1[i]);
  else if (i - n < n)     o2[i - n] = f2bf(w2[i - n]);
}

// ---------------- bf16 WMMA GEMM: C = A(MxK) * B(KxN) + bias, epilogue ------
// MODE 0: SiLU -> bf16 Hout       MODE 1: + Yres -> f32 Fout
#define BM 128
#define BN 128
#define BKK 32
#define LDA 36    // ushorts per A LDS row (32 + 4 pad -> conflict free)
#define LDAW 18   // u32 per A LDS row
#define LDB 17    // u32 per B LDS row (16 + 1 pad -> conflict free)

template <int MODE>
__global__ __launch_bounds__(256) void gemm_bf16_kernel(
    const unsigned short* __restrict__ A, const unsigned short* __restrict__ Bm,
    const float* __restrict__ bias,
    unsigned short* __restrict__ Hout, float* __restrict__ Fout,
    const float* __restrict__ Yres, int M, int N, int K) {
  __shared__ unsigned short As[2][BM * LDA];
  __shared__ unsigned       Bs[2][BN * LDB];

  const int tid = threadIdx.x;
  const int m0  = blockIdx.x * BM;
  const int n0  = blockIdx.y * BN;

  // global-load mapping: 256 threads, each 2x uint4 of A and 2x uint4 of B
  const int arow = tid >> 2;   // 0..63 (second row = +64)
  const int akq  = tid & 3;    // 16B chunk within 64B row-slab
  const int br2  = tid >> 4;   // 0..15 -> K rows 2*br2, 2*br2+1
  const int bc16 = tid & 15;   // col block: bc16*8 .. +7

  const int nkb = K / BKK;
  uint4 aR0, aR1, bR0, bR1;

  auto gload = [&](int kb) {
    const unsigned short* ap0 = A + (size_t)(m0 + arow) * K + kb * BKK;
    const unsigned short* ap1 = A + (size_t)(m0 + arow + 64) * K + kb * BKK;
    aR0 = ((const uint4*)ap0)[akq];
    aR1 = ((const uint4*)ap1)[akq];
    const unsigned short* bp0 = Bm + (size_t)(kb * BKK + 2 * br2) * N + n0 + bc16 * 8;
    bR0 = *(const uint4*)bp0;
    bR1 = *(const uint4*)(bp0 + N);
  };
  auto sstore = [&](int buf) {
    unsigned* ad0 = (unsigned*)&As[buf][(size_t)arow * LDA + akq * 8];
    ad0[0] = aR0.x; ad0[1] = aR0.y; ad0[2] = aR0.z; ad0[3] = aR0.w;
    unsigned* ad1 = (unsigned*)&As[buf][(size_t)(arow + 64) * LDA + akq * 8];
    ad1[0] = aR1.x; ad1[1] = aR1.y; ad1[2] = aR1.z; ad1[3] = aR1.w;
    const unsigned r0[4] = {bR0.x, bR0.y, bR0.z, bR0.w};
    const unsigned r1[4] = {bR1.x, bR1.y, bR1.z, bR1.w};
#pragma unroll
    for (int e = 0; e < 8; ++e) {          // pack K pairs: B[2kp][n] | B[2kp+1][n]<<16
      const unsigned lo = (r0[e >> 1] >> ((e & 1) * 16)) & 0xFFFFu;
      const unsigned hi = (r1[e >> 1] >> ((e & 1) * 16)) & 0xFFFFu;
      Bs[buf][(bc16 * 8 + e) * LDB + br2] = lo | (hi << 16);
    }
  };

  const int lane  = tid & 31;
  const int l16   = lane & 15;
  const int khalf = lane >> 4;
  const int wid   = tid >> 5;
  const int wM    = (wid >> 1) * 32;   // 4 wave-rows
  const int wN    = (wid & 1) * 64;    // 2 wave-cols

  v8f acc[2][4];
#pragma unroll
  for (int mt = 0; mt < 2; ++mt)
#pragma unroll
    for (int nt = 0; nt < 4; ++nt)
#pragma unroll
      for (int i = 0; i < 8; ++i) acc[mt][nt][i] = 0.0f;

  gload(0);
  sstore(0);
  for (int kb = 0; kb < nkb; ++kb) {
    __syncthreads();
    if (kb + 1 < nkb) gload(kb + 1);           // overlap next tile's global loads
    const int buf = kb & 1;
    const unsigned* As32 = (const unsigned*)&As[buf][0];
    const unsigned* Bs32 = &Bs[buf][0];
    Frag af[2], bfr[4];
#pragma unroll
    for (int mt = 0; mt < 2; ++mt) {
      const int m = wM + mt * 16 + l16;
#pragma unroll
      for (int v = 0; v < 8; ++v) {
        // A 16x32 bf16 layout: VGPR v<4 -> kp = v + 4*khalf ; v>=4 -> kp = v+4+4*khalf
        const int kp = (v < 4) ? (v + 4 * khalf) : (v + 4 + 4 * khalf);
        af[mt].u[v] = As32[m * LDAW + kp];
      }
    }
#pragma unroll
    for (int nt = 0; nt < 4; ++nt) {
      const int n = wN + nt * 16 + l16;
#pragma unroll
      for (int v = 0; v < 8; ++v) {
        // B 32x16 bf16 layout: VGPR v<4 -> kp = v + 4*khalf ; v>=4 -> kp = v+4+4*khalf
        const int kp = (v < 4) ? (v + 4 * khalf) : (v + 4 + 4 * khalf);
        bfr[nt].u[v] = Bs32[n * LDB + kp];
      }
    }
#pragma unroll
    for (int mt = 0; mt < 2; ++mt)
#pragma unroll
      for (int nt = 0; nt < 4; ++nt)
        acc[mt][nt] = __builtin_amdgcn_wmma_f32_16x16x32_bf16(
            false, af[mt].v, false, bfr[nt].v, (short)0, acc[mt][nt], false, false);
    if (kb + 1 < nkb) sstore((kb + 1) & 1);
  }

  // epilogue: C/D layout -> element (M = v + 8*khalf, N = l16) per 16x16 tile
#pragma unroll
  for (int mt = 0; mt < 2; ++mt) {
#pragma unroll
    for (int nt = 0; nt < 4; ++nt) {
      const int cc = n0 + wN + nt * 16 + l16;
#pragma unroll
      for (int v = 0; v < 8; ++v) {
        const int rr = m0 + wM + mt * 16 + v + 8 * khalf;
        const float val = acc[mt][nt][v] + bias[cc];
        if (MODE == 0) {
          const float sv = val / (1.0f + __expf(-val));   // SiLU
          Hout[(size_t)rr * N + cc] = f2bf(sv);
        } else {
          Fout[(size_t)rr * N + cc] = val + Yres[(size_t)rr * N + cc];
        }
      }
    }
  }
}

// ---------------- launcher --------------------------------------------------
extern "C" void kernel_launch(void* const* d_in, const int* in_sizes, int n_in,
                              void* d_out, int out_size, void* d_ws, size_t ws_size,
                              hipStream_t stream) {
  (void)in_sizes; (void)n_in; (void)out_size; (void)ws_size;
  const float* x      = (const float*)d_in[0];
  const float* hid_re = (const float*)d_in[1];
  const float* hid_im = (const float*)d_in[2];
  const float* phz_re = (const float*)d_in[3];
  const float* phz_im = (const float*)d_in[4];
  const float* pin_re = (const float*)d_in[5];
  const float* pin_im = (const float*)d_in[6];
  const float* ln_g   = (const float*)d_in[7];
  const float* ln_b   = (const float*)d_in[8];
  const float* w1     = (const float*)d_in[9];
  const float* b1     = (const float*)d_in[10];
  const float* w2     = (const float*)d_in[11];
  const float* b2     = (const float*)d_in[12];

  float*  outp   = (float*)d_out;
  float2* hidOut = (float2*)(outp + (size_t)Bsz * Lsz * Dsz);  // complex64, interleaved

  // workspace bump allocator (256B aligned)
  char* wp = (char*)d_ws;
  auto alloc = [&](size_t bytes) -> void* {
    void* p = wp;
    wp += (bytes + 255) & ~(size_t)255;
    return p;
  };
  float*  mean1    = (float*)alloc((size_t)Rrows * 4);
  float*  rstd1    = (float*)alloc((size_t)Rrows * 4);
  float2* carryLoc = (float2*)alloc((size_t)Bsz * NC * Dsz * 8);
  float2* carryIn  = (float2*)alloc((size_t)Bsz * NC * Dsz * 8);
  float*  y        = (float*)alloc((size_t)Rrows * Dsz * 4);
  unsigned short* yn  = (unsigned short*)alloc((size_t)Rrows * Dsz * 2);
  unsigned short* h   = (unsigned short*)alloc((size_t)Rrows * DFsz * 2);
  unsigned short* w1b = (unsigned short*)alloc((size_t)Dsz * DFsz * 2);
  unsigned short* w2b = (unsigned short*)alloc((size_t)DFsz * Dsz * 2);

  const int nW = Dsz * DFsz;  // 524288 elements each

  // 1) LN1 stats on x
  ln_stats_kernel<<<Rrows / 8, 256, 0, stream>>>(x, mean1, rstd1, Rrows);
  // 2) chunked local scans
  scan_pass1_kernel<<<(Bsz * NC * Dsz) / 256, 256, 0, stream>>>(
      x, mean1, rstd1, ln_g, ln_b, phz_re, phz_im, pin_re, pin_im, carryLoc);
  // 3) carry prefix across chunks
  scan_carry_kernel<<<(Bsz * Dsz) / 256, 256, 0, stream>>>(
      carryLoc, hid_re, hid_im, phz_re, phz_im, carryIn);
  // 4) final scan -> hidden (complex out) and y
  scan_pass2_kernel<<<(Bsz * NC * Dsz) / 256, 256, 0, stream>>>(
      x, mean1, rstd1, ln_g, ln_b, phz_re, phz_im, pin_re, pin_im,
      carryIn, hidOut, y);
  // 5) weights -> bf16
  wcvt_kernel<<<(2 * nW) / 256, 256, 0, stream>>>(w1, w2, w1b, w2b, nW);
  // 6) LN2 fused -> yn (bf16)
  ln_bf16_kernel<<<Rrows / 8, 256, 0, stream>>>(y, ln_g, ln_b, yn, Rrows);
  // 7) GEMM1: h = silu(yn @ w1 + b1)   (16384x1024x512, WMMA bf16)
  gemm_bf16_kernel<0><<<dim3(Rrows / BM, DFsz / BN), 256, 0, stream>>>(
      yn, w1b, b1, h, nullptr, nullptr, Rrows, DFsz, Dsz);
  // 8) GEMM2: out = h @ w2 + b2 + y    (16384x512x1024, WMMA bf16)
  gemm_bf16_kernel<1><<<dim3(Rrows / BM, Dsz / BN), 256, 0, stream>>>(
      h, w2b, b2, nullptr, outp, y, Rrows, Dsz, DFsz);
}
